// GatedFusionBlockCustom_42322607735393
// MI455X (gfx1250) — compile-verified, matrix operand order checked by reading, not run on hardware
//
#include <hip/hip_runtime.h>
#include <hip/hip_bf16.h>

// ---------------------------------------------------------------------------
// GatedFusionBlock on MI455X (gfx1250): bf16 WMMA GEMMs + flash attention,
// with gfx1250 async global->LDS staging (ASYNCcnt) and double buffering.
// B=8, S=2048, H=256, NH=8, DH=32, M = B*S = 16384 tokens.
// ---------------------------------------------------------------------------

typedef __attribute__((ext_vector_type(16))) __bf16 v16bf;
typedef __attribute__((ext_vector_type(8)))  float  v8f;

union FragBF {
    uint4          q[2];
    v16bf          v;
    unsigned short u[16];
};

__device__ __forceinline__ unsigned short f2bf(float f) {
    unsigned int x = __builtin_bit_cast(unsigned int, f);
    unsigned int r = x + 0x7FFFu + ((x >> 16) & 1u);   // round-to-nearest-even
    return (unsigned short)(r >> 16);
}

// gfx1250 async copy: per-lane 16B global -> LDS, tracked by ASYNCcnt.
// Generic pointers to __shared__ carry the LDS byte offset in their low 32 bits.
__device__ __forceinline__ void async_ld_b128(const void* gaddr, void* lds_ptr) {
    unsigned l = (unsigned)(uintptr_t)lds_ptr;
    asm volatile("global_load_async_to_lds_b128 %0, %1, off"
                 :: "v"(l), "v"(gaddr)
                 : "memory");
}
__device__ __forceinline__ void wait_async0() {
    asm volatile("s_wait_asynccnt 0x0" ::: "memory");
}

__device__ __forceinline__ v8f wmma_bf16(const FragBF& a, const FragBF& b, v8f c) {
    return __builtin_amdgcn_wmma_f32_16x16x32_bf16(false, a.v, false, b.v,
                                                   (short)0, c, false, false);
}

// ---------------------------------------------------------------------------
// Means over sequence -> joint[b, 0:256]=video mean, [256:512]=audio mean
// ---------------------------------------------------------------------------
__global__ __launch_bounds__(256)
void mean_kernel(const float* __restrict__ video, const float* __restrict__ audio,
                 float* __restrict__ joint) {
    const int bb = blockIdx.x, h = threadIdx.x;
    const float* vp = video + (size_t)bb * 2048 * 256 + h;
    const float* ap = audio + (size_t)bb * 2048 * 256 + h;
    float sv = 0.f, sa = 0.f;
    for (int s = 0; s < 2048; ++s) {
        sv += vp[(size_t)s * 256];
        sa += ap[(size_t)s * 256];
    }
    joint[bb * 512 + h]       = sv * (1.f / 2048.f);
    joint[bb * 512 + 256 + h] = sa * (1.f / 2048.f);
}

// ---------------------------------------------------------------------------
// gate = tanh(w2 . relu(W1 @ joint + b1) + b2)      (512 -> 128 -> 1)
// ---------------------------------------------------------------------------
__global__ __launch_bounds__(128)
void gate_kernel(const float* __restrict__ joint,
                 const float* __restrict__ w1, const float* __restrict__ b1,
                 const float* __restrict__ w2, const float* __restrict__ b2,
                 float* __restrict__ gout) {
    __shared__ float hs[128];
    const int bb = blockIdx.x, t = threadIdx.x;
    const float* j  = joint + bb * 512;
    const float* wr = w1 + (size_t)t * 512;
    float s = b1[t];
    for (int k = 0; k < 512; ++k) s += j[k] * wr[k];
    hs[t] = fmaxf(s, 0.f) * w2[t];
    __syncthreads();
    for (int off = 64; off >= 1; off >>= 1) {
        if (t < off) hs[t] += hs[t + off];
        __syncthreads();
    }
    if (t == 0) gout[bb] = tanhf(hs[0] + b2[0]);
}

// ---------------------------------------------------------------------------
// Broadcast per-batch gates into output sections 1 and 2
// ---------------------------------------------------------------------------
__global__ void bcast_gates(const float* __restrict__ gm, const float* __restrict__ gf,
                            float* __restrict__ out1, float* __restrict__ out2) {
    size_t i = (size_t)blockIdx.x * blockDim.x + threadIdx.x;
    if (i < (size_t)8 * 2048 * 256) {
        const int bidx = (int)(i >> 19);   // 2048*256 = 2^19
        out1[i] = gm[bidx];
        out2[i] = gf[bidx];
    }
}

// ---------------------------------------------------------------------------
// f32 -> bf16 weight conversion
// ---------------------------------------------------------------------------
__global__ void f2bf_kernel(const float* __restrict__ src,
                            unsigned short* __restrict__ dst, int n) {
    int i = blockIdx.x * blockDim.x + threadIdx.x;
    if (i < n) dst[i] = f2bf(src[i]);
}

// ---------------------------------------------------------------------------
// Combined projection: Wc = outproj_w @ aproj_w (256x256), bc = O@ab + ob
// (x @ A^T) @ O^T + ... == x @ (O A)^T + (O ab + ob)
// ---------------------------------------------------------------------------
__global__ __launch_bounds__(256)
void combine_proj_kernel(const float* __restrict__ O, const float* __restrict__ A,
                         const float* __restrict__ ab, const float* __restrict__ ob,
                         unsigned short* __restrict__ Wc, float* __restrict__ bc) {
    const int i = blockIdx.x, j = threadIdx.x;
    float s = 0.f;
    for (int k = 0; k < 256; ++k) s += O[i * 256 + k] * A[k * 256 + j];
    Wc[i * 256 + j] = f2bf(s);
    if (j == 0) {
        float t = ob[i];
        for (int k = 0; k < 256; ++k) t += O[i * 256 + k] * ab[k];
        bc[i] = t;
    }
}

// ---------------------------------------------------------------------------
// LayerNorm over H=256, one wave per token, output bf16
// ---------------------------------------------------------------------------
__global__ __launch_bounds__(256)
void ln_bf16_kernel(const float* __restrict__ X, const float* __restrict__ g,
                    const float* __restrict__ b, unsigned short* __restrict__ out) {
    const int lane = threadIdx.x & 31;
    const int wv   = threadIdx.x >> 5;
    const int row  = blockIdx.x * 8 + wv;
    const float* xr = X + (size_t)row * 256;
    float x[8], s = 0.f;
    for (int i = 0; i < 8; ++i) { x[i] = xr[lane + 32 * i]; s += x[i]; }
    for (int off = 16; off >= 1; off >>= 1) s += __shfl_xor(s, off, 32);
    const float mu = s * (1.f / 256.f);
    float v = 0.f;
    for (int i = 0; i < 8; ++i) { float d = x[i] - mu; v += d * d; }
    for (int off = 16; off >= 1; off >>= 1) v += __shfl_xor(v, off, 32);
    const float rstd = rsqrtf(v * (1.f / 256.f) + 1e-5f);
    for (int i = 0; i < 8; ++i) {
        const int c = lane + 32 * i;
        out[(size_t)row * 256 + c] = f2bf((x[i] - mu) * rstd * g[c] + b[c]);
    }
}

// ---------------------------------------------------------------------------
// bf16 WMMA GEMM: out = epi( X[M,K] @ W[N,K]^T + bias )
// Block tile 128x64 (8 waves; each wave owns a 32x32 register tile = 4 WMMA
// accumulators). K stepped by 32 through double-buffered LDS filled with
// async global->LDS copies.
// ---------------------------------------------------------------------------
__global__ __launch_bounds__(256)
void gemm_bf16_kernel(const unsigned short* __restrict__ X,
                      const unsigned short* __restrict__ W,
                      const float* __restrict__ bias,
                      const float* __restrict__ gate,     // per-batch, may be null
                      const float* __restrict__ res,      // f32 [M,N], may be null
                      float* __restrict__ outf,           // f32 out (if non-null)
                      unsigned short* __restrict__ outb,  // bf16 out otherwise
                      int M, int N, int K, int relu) {
    __shared__ __align__(16) unsigned short lsA[2][128][40];
    __shared__ __align__(16) unsigned short lsB[2][64][40];

    const int tid  = threadIdx.x;
    const int lane = tid & 31;
    const int wid  = tid >> 5;
    const int hf   = lane >> 4;
    const int l16  = lane & 15;
    const int wr   = wid & 3;    // 4 wave-rows of 32
    const int wc   = wid >> 2;   // 2 wave-cols of 32
    const int m0   = blockIdx.y * 128;
    const int n0   = blockIdx.x * 64;

    const int srow = tid >> 2;   // 0..63
    const int sq   = tid & 3;    // 16B quad within 32-col slab

    v8f acc00 = {}, acc01 = {}, acc10 = {}, acc11 = {};

    // Prologue: fill buffer 0 for kk = 0
    async_ld_b128(X + (size_t)(m0 + srow) * K + sq * 8,       &lsA[0][srow][sq * 8]);
    async_ld_b128(X + (size_t)(m0 + 64 + srow) * K + sq * 8,  &lsA[0][64 + srow][sq * 8]);
    async_ld_b128(W + (size_t)(n0 + srow) * K + sq * 8,       &lsB[0][srow][sq * 8]);

    int cur = 0;
    for (int kk = 0; kk < K; kk += 32) {
        wait_async0();       // this wave's fills (into buf cur) landed
        __syncthreads();     // everyone's fills landed
        if (kk + 32 < K) {   // prefetch next slab into the other buffer
            const int nk = kk + 32, nb = cur ^ 1;
            async_ld_b128(X + (size_t)(m0 + srow) * K + nk + sq * 8,      &lsA[nb][srow][sq * 8]);
            async_ld_b128(X + (size_t)(m0 + 64 + srow) * K + nk + sq * 8, &lsA[nb][64 + srow][sq * 8]);
            async_ld_b128(W + (size_t)(n0 + srow) * K + nk + sq * 8,      &lsB[nb][srow][sq * 8]);
        }
        // Fragment gather: element j<8 -> k=j+hf*8 ; j>=8 -> k=16+(j-8)+hf*8
        FragBF a0, a1, b0, b1;
        a0.q[0] = *(const uint4*)&lsA[cur][wr * 32 + l16][hf * 8];
        a0.q[1] = *(const uint4*)&lsA[cur][wr * 32 + l16][16 + hf * 8];
        a1.q[0] = *(const uint4*)&lsA[cur][wr * 32 + 16 + l16][hf * 8];
        a1.q[1] = *(const uint4*)&lsA[cur][wr * 32 + 16 + l16][16 + hf * 8];
        b0.q[0] = *(const uint4*)&lsB[cur][wc * 32 + l16][hf * 8];
        b0.q[1] = *(const uint4*)&lsB[cur][wc * 32 + l16][16 + hf * 8];
        b1.q[0] = *(const uint4*)&lsB[cur][wc * 32 + 16 + l16][hf * 8];
        b1.q[1] = *(const uint4*)&lsB[cur][wc * 32 + 16 + l16][16 + hf * 8];
        acc00 = wmma_bf16(a0, b0, acc00);
        acc01 = wmma_bf16(a0, b1, acc01);
        acc10 = wmma_bf16(a1, b0, acc10);
        acc11 = wmma_bf16(a1, b1, acc11);
        cur ^= 1;
    }

    // Epilogue. C layout: VGPR r -> row = +hf*8 + r, col = +l16
    #pragma unroll
    for (int ct = 0; ct < 2; ++ct) {
        const int col   = n0 + wc * 32 + ct * 16 + l16;
        const float bcl = bias ? bias[col] : 0.f;
        #pragma unroll
        for (int rt = 0; rt < 2; ++rt) {
            const v8f acc = (rt == 0) ? (ct == 0 ? acc00 : acc01)
                                      : (ct == 0 ? acc10 : acc11);
            #pragma unroll
            for (int r = 0; r < 8; ++r) {
                const int row = m0 + wr * 32 + rt * 16 + hf * 8 + r;
                float v = acc[r] + bcl;
                if (gate) v *= gate[row >> 11];          // 2048 tokens per batch
                if (res)  v += res[(size_t)row * N + col];
                if (outf) {
                    outf[(size_t)row * N + col] = v;
                } else {
                    if (relu) v = fmaxf(v, 0.f);
                    outb[(size_t)row * N + col] = f2bf(v);
                }
            }
        }
    }
}

// ---------------------------------------------------------------------------
// Flash attention, bf16 WMMA. Block = (qtile of 128 rows, head, batch).
// qkv: bf16 [M, 768] (q | k | v, 32 per head). ctx: bf16 [M, 256].
// K/V tiles (32 keys x 32 dh) double-buffered via async global->LDS.
// ---------------------------------------------------------------------------
__global__ __launch_bounds__(256)
void attn_kernel(const unsigned short* __restrict__ qkv,
                 unsigned short* __restrict__ ctx) {
    __shared__ __align__(16) unsigned short lsK[2][32][40];
    __shared__ __align__(16) unsigned short lsV[2][32][40];
    __shared__ __align__(16) unsigned short lsP[8][16][40];

    const int tid  = threadIdx.x;
    const int lane = tid & 31;
    const int wid  = tid >> 5;
    const int hf   = lane >> 4;
    const int l16  = lane & 15;
    const int qt   = blockIdx.x, h = blockIdx.y, bb = blockIdx.z;
    const float scale = 0.17677669529663687f;   // 1/sqrt(32)

    // Q fragment (16 rows x 32 dh) -- loaded once
    const int    qrow = qt * 128 + wid * 16 + l16;
    const size_t qtok = (size_t)bb * 2048 + qrow;
    const unsigned short* qp = qkv + qtok * 768 + h * 32;
    FragBF qa;
    qa.q[0] = *(const uint4*)(qp + hf * 8);
    qa.q[1] = *(const uint4*)(qp + 16 + hf * 8);

    float mrow[8], lrow[8];
    v8f o0 = {}, o1 = {};
    for (int r = 0; r < 8; ++r) { mrow[r] = -3.0e38f; lrow[r] = 0.f; }

    const int sr = (tid & 127) >> 2, sq = tid & 3;
    // Prologue fill: waves 0-3 stage K, waves 4-7 stage V (1 async 16B each)
    {
        const size_t tok = (size_t)bb * 2048 + sr;
        if (tid < 128)
            async_ld_b128(qkv + tok * 768 + 256 + h * 32 + sq * 8, &lsK[0][sr][sq * 8]);
        else
            async_ld_b128(qkv + tok * 768 + 512 + h * 32 + sq * 8, &lsV[0][sr][sq * 8]);
    }

    int cur = 0;
    for (int kb = 0; kb < 2048; kb += 32) {
        wait_async0();
        __syncthreads();
        if (kb + 32 < 2048) {
            const int nb = cur ^ 1;
            const size_t tok = (size_t)bb * 2048 + kb + 32 + sr;
            if (tid < 128)
                async_ld_b128(qkv + tok * 768 + 256 + h * 32 + sq * 8, &lsK[nb][sr][sq * 8]);
            else
                async_ld_b128(qkv + tok * 768 + 512 + h * 32 + sq * 8, &lsV[nb][sr][sq * 8]);
        }

        // Scores: S = Q (16x32dh) x K^T for two 16-key tiles
        FragBF k0, k1;
        k0.q[0] = *(const uint4*)&lsK[cur][l16][hf * 8];
        k0.q[1] = *(const uint4*)&lsK[cur][l16][16 + hf * 8];
        k1.q[0] = *(const uint4*)&lsK[cur][16 + l16][hf * 8];
        k1.q[1] = *(const uint4*)&lsK[cur][16 + l16][16 + hf * 8];
        v8f z = {};
        v8f s0 = wmma_bf16(qa, k0, z);
        v8f s1 = wmma_bf16(qa, k1, z);

        // Online softmax over this 32-key slab (row data lives in one 16-lane half)
        float p0[8], p1[8];
        #pragma unroll
        for (int r = 0; r < 8; ++r) {
            const float a = s0[r] * scale, b = s1[r] * scale;
            float t = fmaxf(a, b);
            t = fmaxf(t, __shfl_xor(t, 1, 32));
            t = fmaxf(t, __shfl_xor(t, 2, 32));
            t = fmaxf(t, __shfl_xor(t, 4, 32));
            t = fmaxf(t, __shfl_xor(t, 8, 32));
            const float mn    = fmaxf(mrow[r], t);
            const float alpha = __expf(mrow[r] - mn);
            mrow[r] = mn;
            p0[r] = __expf(a - mn);
            p1[r] = __expf(b - mn);
            float rs = p0[r] + p1[r];
            rs += __shfl_xor(rs, 1, 32);
            rs += __shfl_xor(rs, 2, 32);
            rs += __shfl_xor(rs, 4, 32);
            rs += __shfl_xor(rs, 8, 32);
            lrow[r] = lrow[r] * alpha + rs;
            o0[r] *= alpha;
            o1[r] *= alpha;
        }

        // C-layout -> A-layout for P via wave-private LDS (LDS ops are in-order)
        #pragma unroll
        for (int r = 0; r < 8; ++r) {
            const int pr = hf * 8 + r;
            lsP[wid][pr][l16]      = f2bf(p0[r]);
            lsP[wid][pr][16 + l16] = f2bf(p1[r]);
        }
        FragBF pa;
        pa.q[0] = *(const uint4*)&lsP[wid][l16][hf * 8];
        pa.q[1] = *(const uint4*)&lsP[wid][l16][16 + hf * 8];

        // V fragments: B is (32 keys x 16 dh); element j -> key index
        FragBF v0, v1;
        #pragma unroll
        for (int j = 0; j < 16; ++j) {
            const int key = (j < 8) ? (j + hf * 8) : (16 + (j - 8) + hf * 8);
            v0.u[j] = lsV[cur][key][l16];
            v1.u[j] = lsV[cur][key][16 + l16];
        }
        o0 = wmma_bf16(pa, v0, o0);
        o1 = wmma_bf16(pa, v1, o1);
        cur ^= 1;
    }

    #pragma unroll
    for (int r = 0; r < 8; ++r) {
        const int    row = qt * 128 + wid * 16 + hf * 8 + r;
        const size_t tok = (size_t)bb * 2048 + row;
        const float  inv = 1.f / lrow[r];
        ctx[tok * 256 + h * 32 + l16]      = f2bf(o0[r] * inv);
        ctx[tok * 256 + h * 32 + 16 + l16] = f2bf(o1[r] * inv);
    }
}

// ---------------------------------------------------------------------------
// Host launcher
// ---------------------------------------------------------------------------
extern "C" void kernel_launch(void* const* d_in, const int* in_sizes, int n_in,
                              void* d_out, int out_size, void* d_ws, size_t ws_size,
                              hipStream_t stream) {
    const int M = 8 * 2048;             // tokens
    const size_t SEC = (size_t)M * 256; // one output section

    const float* video     = (const float*)d_in[0];
    const float* audio     = (const float*)d_in[1];
    const float* g_mha_w1  = (const float*)d_in[2];
    const float* g_mha_b1  = (const float*)d_in[3];
    const float* g_mha_w2  = (const float*)d_in[4];
    const float* g_mha_b2  = (const float*)d_in[5];
    const float* g_ffn_w1  = (const float*)d_in[6];
    const float* g_ffn_b1  = (const float*)d_in[7];
    const float* g_ffn_w2  = (const float*)d_in[8];
    const float* g_ffn_b2  = (const float*)d_in[9];
    const float* aproj_w   = (const float*)d_in[10];
    const float* aproj_b   = (const float*)d_in[11];
    const float* outproj_w = (const float*)d_in[12];
    const float* outproj_b = (const float*)d_in[13];
    const float* ffn1_w1   = (const float*)d_in[14];
    const float* ffn1_b1   = (const float*)d_in[15];
    const float* ffn1_w2   = (const float*)d_in[16];
    const float* ffn1_b2   = (const float*)d_in[17];
    const float* ffn2_w1   = (const float*)d_in[18];
    const float* ffn2_b1   = (const float*)d_in[19];
    const float* ffn2_w2   = (const float*)d_in[20];
    const float* ffn2_b2   = (const float*)d_in[21];
    const float* attn_in_w = (const float*)d_in[22];
    const float* attn_in_b = (const float*)d_in[23];
    const float* attn_out_w= (const float*)d_in[24];
    const float* attn_out_b= (const float*)d_in[25];
    const float* n1_g = (const float*)d_in[26]; const float* n1_b = (const float*)d_in[27];
    const float* n2_g = (const float*)d_in[28]; const float* n2_b = (const float*)d_in[29];
    const float* n3_g = (const float*)d_in[30]; const float* n3_b = (const float*)d_in[31];
    const float* n4_g = (const float*)d_in[32]; const float* n4_b = (const float*)d_in[33];

    // Workspace carve-out (256B aligned)
    char* ws = (char*)d_ws;
    size_t off = 0;
    auto alloc = [&](size_t bytes) -> char* {
        char* p = ws + off;
        off = (off + bytes + 255) & ~(size_t)255;
        return p;
    };
    float*          joint = (float*)alloc(8 * 512 * 4);
    float*          gm    = (float*)alloc(8 * 4);
    float*          gf    = (float*)alloc(8 * 4);
    unsigned short* Wc    = (unsigned short*)alloc(256 * 256 * 2);
    float*          bc    = (float*)alloc(256 * 4);
    unsigned short* wF1a  = (unsigned short*)alloc(1024 * 256 * 2);
    unsigned short* wF1b  = (unsigned short*)alloc(256 * 1024 * 2);
    unsigned short* wF2a  = (unsigned short*)alloc(1024 * 256 * 2);
    unsigned short* wF2b  = (unsigned short*)alloc(256 * 1024 * 2);
    unsigned short* wQKV  = (unsigned short*)alloc(768 * 256 * 2);
    unsigned short* wAO   = (unsigned short*)alloc(256 * 256 * 2);
    unsigned short* tokA  = (unsigned short*)alloc((size_t)M * 256 * 2);  // LN outputs
    unsigned short* tokB  = (unsigned short*)alloc((size_t)M * 256 * 2);  // ctx
    unsigned short* hid   = (unsigned short*)alloc((size_t)M * 1024 * 2); // hidden / qkv
    float*          Z     = (float*)alloc((size_t)M * 256 * 4);           // z, later refined_z
    float*          Zb    = (float*)alloc((size_t)M * 256 * 4);           // z_bar

    float* out_final = (float*)d_out;
    float* out_gm    = out_final + SEC;
    float* out_gf    = out_final + 2 * SEC;

    // --- gates ---
    mean_kernel<<<8, 256, 0, stream>>>(video, audio, joint);
    gate_kernel<<<8, 128, 0, stream>>>(joint, g_mha_w1, g_mha_b1, g_mha_w2, g_mha_b2, gm);
    gate_kernel<<<8, 128, 0, stream>>>(joint, g_ffn_w1, g_ffn_b1, g_ffn_w2, g_ffn_b2, gf);
    bcast_gates<<<(int)((SEC + 255) / 256), 256, 0, stream>>>(gm, gf, out_gm, out_gf);

    // --- weight prep ---
    auto cvt = [&](const float* s, unsigned short* d, int n) {
        f2bf_kernel<<<(n + 255) / 256, 256, 0, stream>>>(s, d, n);
    };
    cvt(ffn1_w1, wF1a, 1024 * 256);
    cvt(ffn1_w2, wF1b, 256 * 1024);
    cvt(ffn2_w1, wF2a, 1024 * 256);
    cvt(ffn2_w2, wF2b, 256 * 1024);
    cvt(attn_in_w, wQKV, 768 * 256);
    cvt(attn_out_w, wAO, 256 * 256);
    combine_proj_kernel<<<256, 256, 0, stream>>>(outproj_w, aproj_w, aproj_b, outproj_b, Wc, bc);

    const dim3 blk(256);
    const dim3 g256(256 / 64, M / 128);   // N=256
    const dim3 g768(768 / 64, M / 128);   // N=768
    const dim3 g1024(1024 / 64, M / 128); // N=1024

    // z = gm * (LN1(audio) @ Wc^T + bc) + video
    ln_bf16_kernel<<<M / 8, 256, 0, stream>>>(audio, n1_g, n1_b, tokA);
    gemm_bf16_kernel<<<g256, blk, 0, stream>>>(tokA, Wc, bc, gm, video, Z, nullptr,
                                               M, 256, 256, 0);
    // z_bar = gf * FFN1(LN2(z)) + z
    ln_bf16_kernel<<<M / 8, 256, 0, stream>>>(Z, n2_g, n2_b, tokA);
    gemm_bf16_kernel<<<g1024, blk, 0, stream>>>(tokA, wF1a, ffn1_b1, nullptr, nullptr,
                                                nullptr, hid, M, 1024, 256, 1);
    gemm_bf16_kernel<<<g256, blk, 0, stream>>>(hid, wF1b, ffn1_b2, gf, Z, Zb, nullptr,
                                               M, 256, 1024, 0);
    // self-attention on LN3(z_bar)
    ln_bf16_kernel<<<M / 8, 256, 0, stream>>>(Zb, n3_g, n3_b, tokA);
    gemm_bf16_kernel<<<g768, blk, 0, stream>>>(tokA, wQKV, attn_in_b, nullptr, nullptr,
                                               nullptr, hid, M, 768, 256, 0);
    attn_kernel<<<dim3(16, 8, 8), blk, 0, stream>>>(hid, tokB);
    // refined_z = ctx @ Wo^T + bo + z_bar   (reuse Z buffer)
    gemm_bf16_kernel<<<g256, blk, 0, stream>>>(tokB, wAO, attn_out_b, nullptr, Zb, Z,
                                               nullptr, M, 256, 256, 0);
    // final = FFN2(LN4(refined_z)) + refined_z
    ln_bf16_kernel<<<M / 8, 256, 0, stream>>>(Z, n4_g, n4_b, tokA);
    gemm_bf16_kernel<<<g1024, blk, 0, stream>>>(tokA, wF2a, ffn2_b1, nullptr, nullptr,
                                                nullptr, hid, M, 1024, 256, 1);
    gemm_bf16_kernel<<<g256, blk, 0, stream>>>(hid, wF2b, ffn2_b2, nullptr, Z, out_final,
                                               nullptr, M, 256, 1024, 0);
    (void)in_sizes; (void)n_in; (void)out_size; (void)ws_size;
}
